// Gradient3DHMM_4346506903765
// MI455X (gfx1250) — compile-verified, matrix operand order checked by reading, not run on hardware
//
#include <hip/hip_runtime.h>
#include <hip/hip_bf16.h>

#define XYD 16
#define SS  4096
#define VV  2048
#define BB  8
#define WW  8

typedef __attribute__((ext_vector_type(16))) _Float16 v16h;
typedef __attribute__((ext_vector_type(8)))  _Float16 v8h;
typedef __attribute__((ext_vector_type(8)))  float    v8f;

__device__ __forceinline__ int imin(int a, int b) { return a < b ? a : b; }
__device__ __forceinline__ int imax(int a, int b) { return a > b ? a : b; }

// ---------- K1: per-row max and log-sum-exp of emis_w ----------
__global__ void k_row_stats(const float* __restrict__ emis, float* __restrict__ stats) {
    __shared__ float red[256];
    int s = blockIdx.x;
    int tid = threadIdx.x;
    const float* row = emis + (size_t)s * VV;
    float m = -INFINITY;
    for (int v = tid; v < VV; v += 256) m = fmaxf(m, row[v]);
    red[tid] = m; __syncthreads();
    for (int off = 128; off > 0; off >>= 1) {
        if (tid < off) red[tid] = fmaxf(red[tid], red[tid + off]);
        __syncthreads();
    }
    m = red[0]; __syncthreads();
    float sum = 0.f;
    for (int v = tid; v < VV; v += 256) sum += __expf(row[v] - m);
    red[tid] = sum; __syncthreads();
    for (int off = 128; off > 0; off >>= 1) {
        if (tid < off) red[tid] += red[tid + off];
        __syncthreads();
    }
    if (tid == 0) { stats[2 * s] = m; stats[2 * s + 1] = __logf(red[0]); }
}

// ---------- K2: 5-point clamped stencil logsumexp; writes f32 and f16 copies ----------
__global__ void k_smooth(const float* __restrict__ emis, const float* __restrict__ stats,
                         float* __restrict__ em, _Float16* __restrict__ em_h) {
    int idx = blockIdx.x * blockDim.x + threadIdx.x;
    if (idx >= SS * VV) return;
    int s = idx >> 11;            // / VV
    int v = idx & (VV - 1);
    int x = s & 15, y = (s >> 4) & 15, z = s >> 8;
    int nn[5];
    nn[0] = s;
    nn[1] = (z << 8) | (imin(y + 1, 15) << 4) | x;   // "up"
    nn[2] = (z << 8) | (imax(y - 1, 0) << 4) | x;    // "dn"
    nn[3] = (z << 8) | (y << 4) | imin(x + 1, 15);   // "lf"
    nn[4] = (z << 8) | (y << 4) | imax(x - 1, 0);    // "rt"
    float vals[5];
    float mx = -INFINITY;
    #pragma unroll
    for (int i = 0; i < 5; ++i) {
        int r = nn[i];
        vals[i] = emis[(size_t)r * VV + v] - stats[2 * r] - stats[2 * r + 1];
        mx = fmaxf(mx, vals[i]);
    }
    float acc = 0.f;
    #pragma unroll
    for (int i = 0; i < 5; ++i) acc += __expf(vals[i] - mx);
    float r = mx + __logf(acc) - 1.6094379124341003f;   // - log 5
    em[idx]   = r;
    em_h[idx] = (_Float16)r;
}

// ---------- K3: build one-hot count matrix (f32 for atomics) ----------
__global__ void k_zero(float* __restrict__ p, int n) {
    int i = blockIdx.x * blockDim.x + threadIdx.x;
    if (i < n) p[i] = 0.f;
}

__global__ void k_scatter(const int* __restrict__ stories, float* __restrict__ counts, int L) {
    int tid = blockIdx.x * blockDim.x + threadIdx.x;
    int total = BB * L * WW;
    if (tid >= total) return;
    int t = (tid / WW) % L;
    int b = tid / (WW * L);
    int token = stories[tid];                        // stories is (B, L, W) row-major
    if (token < VV) atomicAdd(&counts[(size_t)(t * BB + b) * VV + token], 1.0f);
}

__global__ void k_counts_f16(const float* __restrict__ counts, _Float16* __restrict__ counts_h,
                             int n) {
    int i = blockIdx.x * blockDim.x + threadIdx.x;
    if (i < n) counts_h[i] = (_Float16)counts[i];
}

// ---------- K4: allE = counts(M x V) * em(S x V)^T  via WMMA f16->f32 ----------
__global__ void k_gemm_wmma(const _Float16* __restrict__ counts_h,
                            const _Float16* __restrict__ em_h,
                            float* __restrict__ allE, int Mpad) {
    int wave = (int)((blockIdx.x * blockDim.x + threadIdx.x) >> 5);
    int lane = threadIdx.x & 31;
    const int tilesN = SS / 16;                      // 256 N-tiles
    int m0 = (wave / tilesN) * 16;
    int n0 = (wave % tilesN) * 16;
    if (m0 >= Mpad) return;                          // wave-uniform: EXEC stays all-ones
    int half = lane >> 4;
    int l15  = lane & 15;
    // A: 16x32 f16, ISA layout: lane-half g holds K runs [8g..8g+7] and [16+8g..23+8g]
    const _Float16* ap = counts_h + (size_t)(m0 + l15) * VV + 8 * half;
    // B: 32x16, element (k, n) = em[n*VV + k]; same per-lane K runs, lane->column
    const _Float16* bp = em_h + (size_t)(n0 + l15) * VV + 8 * half;
    v8f acc = {};
    for (int k0 = 0; k0 < VV; k0 += 64) {            // K-unroll x2: two WMMAs / iter
        __builtin_prefetch(ap + 128, 0, 1);          // global_prefetch_b8
        __builtin_prefetch(bp + 128, 0, 1);
        v8h a0 = *(const v8h*)(ap);      v8h a1 = *(const v8h*)(ap + 16);
        v8h b0 = *(const v8h*)(bp);      v8h b1 = *(const v8h*)(bp + 16);
        v8h a2 = *(const v8h*)(ap + 32); v8h a3 = *(const v8h*)(ap + 48);
        v8h b2 = *(const v8h*)(bp + 32); v8h b3 = *(const v8h*)(bp + 48);
        v16h a, b, c, d;
        #pragma unroll
        for (int e = 0; e < 8; ++e) {
            a[e] = a0[e]; a[8 + e] = a1[e];
            b[e] = b0[e]; b[8 + e] = b1[e];
            c[e] = a2[e]; c[8 + e] = a3[e];
            d[e] = b2[e]; d[8 + e] = b3[e];
        }
        // 8 args: (neg_a, A, neg_b, B, c_mod, C, reuse_a, reuse_b)
        acc = __builtin_amdgcn_wmma_f32_16x16x32_f16(false, a, false, b, (short)0, acc,
                                                     false, false);
        acc = __builtin_amdgcn_wmma_f32_16x16x32_f16(false, c, false, d, (short)0, acc,
                                                     false, false);
        ap += 64; bp += 64;
    }
    // C/D layout: VGPR r -> M=r (lanes 0-15) / M=r+8 (lanes 16-31), N = lane&15
    #pragma unroll
    for (int r = 0; r < 8; ++r) {
        int m = r + (half ? 8 : 0);
        allE[(size_t)(m0 + m) * SS + n0 + l15] = acc[r];
    }
}

// ---------- K5: sparse transition precompute ----------
__global__ void k_trans(const float* __restrict__ trans_w, float* __restrict__ P,
                        float* __restrict__ amax, int* __restrict__ tgt) {
    int j = blockIdx.x * blockDim.x + threadIdx.x;
    if (j >= SS) return;
    int x = j & 15, y = (j >> 4) & 15, z = j >> 8;
    const int ox[7] = {0, 1, -1, 0, 0, 0, 0};
    const int oy[7] = {0, 0, 0, 1, -1, 0, 0};
    const int oz[7] = {0, 0, 0, 0, 0, 1, 2};
    int nv = 0; int tg[7]; float w[7];
    #pragma unroll
    for (int k = 0; k < 7; ++k) {
        int nx = x + ox[k], ny = y + oy[k], nz = z + oz[k];
        if (nx >= 0 && nx < 16 && ny >= 0 && ny < 16 && nz >= 0 && nz < 16) {
            tg[nv] = nx + (ny << 4) + (nz << 8);
            w[nv]  = trans_w[j * 7 + nv];            // valid-first stable order
            ++nv;
        }
    }
    float wmax = -INFINITY;
    for (int k = 0; k < nv; ++k) wmax = fmaxf(wmax, w[k]);
    float ssum = 0.f;
    for (int k = 0; k < nv; ++k) ssum += __expf(w[k] - wmax);
    float lse = wmax + __logf(ssum);
    amax[j] = wmax - lse;                            // row max of log-probs
    #pragma unroll
    for (int k = 0; k < 7; ++k) {
        if (k < nv) { P[j * 7 + k] = __expf(w[k] - wmax); tgt[j * 7 + k] = tg[k]; }
        else        { P[j * 7 + k] = 0.f;                 tgt[j * 7 + k] = 0;     }
    }
}

// ---------- K6: sequential forward scan (single workgroup, score in ws) ----------
__global__ void __launch_bounds__(1024)
k_forward(const float* __restrict__ allE, const float* __restrict__ P,
          const float* __restrict__ amax, const int* __restrict__ tgt,
          const float* __restrict__ prior_w, float* __restrict__ score,
          float* __restrict__ out, int L) {
    __shared__ float red[1024];
    __shared__ float smax_sh[BB];
    int tid = threadIdx.x;

    // prior log-softmax denominator
    float m = -INFINITY;
    for (int s = tid; s < SS; s += 1024) m = fmaxf(m, prior_w[s]);
    red[tid] = m; __syncthreads();
    for (int off = 512; off > 0; off >>= 1) {
        if (tid < off) red[tid] = fmaxf(red[tid], red[tid + off]);
        __syncthreads();
    }
    float pmax = red[0]; __syncthreads();
    float sum = 0.f;
    for (int s = tid; s < SS; s += 1024) sum += __expf(prior_w[s] - pmax);
    red[tid] = sum; __syncthreads();
    for (int off = 512; off > 0; off >>= 1) {
        if (tid < off) red[tid] += red[tid + off];
        __syncthreads();
    }
    float plse = pmax + __logf(red[0]); __syncthreads();

    // t = 0: score0 = allE[0] + priors
    for (int k = 0; k < 32; ++k) {
        int idx = tid + (k << 10);                   // 0 .. 32767, == b*SS + s
        int b = idx >> 12, s = idx & (SS - 1);
        float v0 = allE[(size_t)b * SS + s] + prior_w[s] - plse;
        score[idx] = v0;
        out[(size_t)b * SS + s] = v0;
    }
    __threadfence(); __syncthreads();

    for (int t = 1; t < L; ++t) {
        // per-batch max of score
        for (int b = 0; b < BB; ++b) {
            float mm = -INFINITY;
            for (int s = tid; s < SS; s += 1024) mm = fmaxf(mm, score[b * SS + s]);
            red[tid] = mm; __syncthreads();
            for (int off = 512; off > 0; off >>= 1) {
                if (tid < off) red[tid] = fmaxf(red[tid], red[tid + off]);
                __syncthreads();
            }
            if (tid == 0) smax_sh[b] = red[0];
            __syncthreads();
        }
        float nv[32];
        for (int k = 0; k < 32; ++k) {
            int idx = tid + (k << 10);
            int b = idx >> 12, j = idx & (SS - 1);
            float sm = smax_sh[b];
            float acc = 0.f;
            #pragma unroll
            for (int q = 0; q < 7; ++q) {
                float p = P[j * 7 + q];
                int  tj = tgt[j * 7 + q];
                acc += p * __expf(score[b * SS + tj] - sm);
            }
            nv[k] = allE[((size_t)t * BB + b) * SS + j] + __logf(acc) + sm + amax[j];
        }
        __syncthreads();                             // all old-score reads done
        for (int k = 0; k < 32; ++k) {
            int idx = tid + (k << 10);
            int b = idx >> 12, j = idx & (SS - 1);
            score[idx] = nv[k];
            out[((size_t)t * BB + b) * SS + j] = nv[k];
        }
        __threadfence(); __syncthreads();
    }
}

extern "C" void kernel_launch(void* const* d_in, const int* in_sizes, int n_in,
                              void* d_out, int out_size, void* d_ws, size_t ws_size,
                              hipStream_t stream) {
    const int*   stories = (const int*)d_in[0];
    const float* trans_w = (const float*)d_in[2];
    const float* emis_w  = (const float*)d_in[3];
    const float* prior_w = (const float*)d_in[4];
    int L    = in_sizes[0] / (BB * WW);
    int M    = L * BB;
    int Mpad = (M + 15) & ~15;

    float* ws = (float*)d_ws;
    size_t o = 0;
    float*    em       = ws + o;              o += (size_t)SS * VV;        // 32 MB
    float*    counts   = ws + o;              o += (size_t)Mpad * VV;      // 2 MB
    float*    allE     = ws + o;              o += (size_t)Mpad * SS;      // 4 MB
    float*    stats    = ws + o;              o += 2 * SS;
    float*    Pm       = ws + o;              o += 7 * SS;
    float*    amax     = ws + o;              o += SS;
    float*    score    = ws + o;              o += (size_t)BB * SS;
    int*      tgt      = (int*)(ws + o);      o += 7 * SS + SS;            // keep 8-elem align
    _Float16* em_h     = (_Float16*)(ws + o); o += (size_t)SS * VV / 2;    // 16 MB
    _Float16* counts_h = (_Float16*)(ws + o);                              // Mpad*VV halves

    k_row_stats<<<SS, 256, 0, stream>>>(emis_w, stats);
    k_smooth<<<(SS * VV) / 256, 256, 0, stream>>>(emis_w, stats, em, em_h);

    int ncnt = Mpad * VV;
    k_zero<<<(ncnt + 255) / 256, 256, 0, stream>>>(counts, ncnt);
    k_scatter<<<(BB * L * WW + 255) / 256, 256, 0, stream>>>(stories, counts, L);
    k_counts_f16<<<(ncnt + 255) / 256, 256, 0, stream>>>(counts, counts_h, ncnt);

    int tiles = (Mpad / 16) * (SS / 16);                 // one wave per 16x16 tile
    k_gemm_wmma<<<(tiles * 32 + 255) / 256, 256, 0, stream>>>(counts_h, em_h, allE, Mpad);

    k_trans<<<SS / 256, 256, 0, stream>>>(trans_w, Pm, amax, tgt);
    k_forward<<<1, 1024, 0, stream>>>(allE, Pm, amax, tgt, prior_w, score, (float*)d_out, L);
}
